// TFT5Attention_5085241279236
// MI455X (gfx1250) — matrix operand order repaired and checked by reading, not compiled, not run on hardware
//
#include <hip/hip_runtime.h>
#include <hip/hip_bf16.h>

typedef __attribute__((ext_vector_type(16))) __bf16 bf16x16;
typedef __attribute__((ext_vector_type(8)))  __bf16 bf16x8;
typedef __attribute__((ext_vector_type(4)))  __bf16 bf16x4;
typedef __attribute__((ext_vector_type(8)))  float  f32x8;

#define DM  1024
#define SEQ 2048
#define NH  16
#define HD  64
#define NB  2

// ---------------- WMMA fragment loaders (CDNA5 wave32 layouts) ----------------
// A (16x32 bf16): lane<16 holds row M=lane, K {0..7,16..23}; lane>=16 K {8..15,24..31}
static __device__ __forceinline__ bf16x16 load_a_frag(const __bf16* p, int half) {
  const __bf16* p0 = p + half * 8;
  bf16x8 lo = *(const bf16x8*)(p0);
  bf16x8 hi = *(const bf16x8*)(p0 + 16);
  bf16x16 a;
#pragma unroll
  for (int j = 0; j < 8; ++j) { a[j] = lo[j]; a[j + 8] = hi[j]; }
  return a;
}
// B (32x16 bf16): lane holds column N=lane%16, K {0..7,16..23} for lane<16; K {8..15,24..31} for lane>=16
static __device__ __forceinline__ bf16x16 load_b_frag(const __bf16* p, int half) {
  const __bf16* p0 = p + half * 8;
  bf16x8 lo = *(const bf16x8*)(p0);
  bf16x8 hi = *(const bf16x8*)(p0 + 16);
  bf16x16 b;
#pragma unroll
  for (int j = 0; j < 8; ++j) { b[j] = lo[j]; b[j + 8] = hi[j]; }
  return b;
}

#define WMMA_BF16(A, B, C) \
  __builtin_amdgcn_wmma_f32_16x16x32_bf16(false, (A), false, (B), (short)0, (C), false, false)

// ---------------- relative-position-bucket LUT: lut[h][rel+2047] ----------------
__global__ void build_lut_kernel(const float* __restrict__ rel_bias, float* __restrict__ lut) {
  int t = blockIdx.x * blockDim.x + threadIdx.x;   // 16*4096 threads
  if (t >= NH * 4096) return;
  int h = t >> 12, x = t & 4095;
  if (x > 4094) { lut[t] = 0.f; return; }
  int rel = x - 2047;      // rel = k - q
  int n = -rel;            // n = q - k
  int ret = 0;
  if (n < 0) { ret = 16; n = -n; }
  int bucket;
  if (n < 8) bucket = n;
  else {
    int vi = 8 + (int)(__logf((float)n * 0.125f) * (1.0f / __logf(16.0f)) * 8.0f);
    bucket = vi < 15 ? vi : 15;
  }
  bucket += ret;
  lut[t] = rel_bias[bucket * NH + h];
}

// position_bias[h][q][k] = lut[h][k - q + 2047]  (vectorized x4, fully coalesced store)
__global__ void write_bias_kernel(const float* __restrict__ lut, float* __restrict__ dst) {
  size_t t = (size_t)blockIdx.x * blockDim.x + threadIdx.x;   // 16.7M threads
  size_t base = t * 4;
  int k = (int)(base & (SEQ - 1));
  int q = (int)((base >> 11) & (SEQ - 1));
  int h = (int)(base >> 22);
  const float* lrow = lut + (size_t)h * 4096 + 2047 - q + k;
  float4 v = make_float4(lrow[0], lrow[1], lrow[2], lrow[3]);
  *(float4*)(dst + base) = v;
}

// ---------------- precision/layout prep ----------------
__global__ void transpose_w_kernel(const float* __restrict__ w, __bf16* __restrict__ wT) {
  int t = blockIdx.x * blockDim.x + threadIdx.x;   // 1M threads
  int n = t >> 10, k = t & (DM - 1);
  wT[t] = (__bf16)w[(size_t)k * DM + n];           // wT[n][k], coalesced store
}

__global__ void cvt_x_kernel(const float* __restrict__ src, __bf16* __restrict__ dst, int n4) {
  int t = blockIdx.x * blockDim.x + threadIdx.x;
  if (t >= n4) return;
  float4 v = ((const float4*)src)[t];
  bf16x4 o; o[0] = (__bf16)v.x; o[1] = (__bf16)v.y; o[2] = (__bf16)v.z; o[3] = (__bf16)v.w;
  ((bf16x4*)dst)[t] = o;
}

// ---------------- QKV projection GEMM: xb[4096x1024] @ wT -> Q/K/Vt (bf16) ----------------
// one wave -> 16x64 output tile; mode: 0=Q [b,h,s,d], 1=K [b,h,s,d], 2=V transposed [b,h,d,s]
__global__ __launch_bounds__(32) void gemm_qkv_kernel(
    const __bf16* __restrict__ A, const __bf16* __restrict__ BT,
    __bf16* __restrict__ dst, int mode) {
  const int lane = threadIdx.x, half = lane >> 4, ml = lane & 15;
  const int m0 = blockIdx.x * 16, n0 = blockIdx.y * 64;

  f32x8 acc[4] = {};
  const __bf16* arow = A + (size_t)(m0 + ml) * DM;
  const __bf16* bcol[4];
#pragma unroll
  for (int g = 0; g < 4; ++g) bcol[g] = BT + (size_t)(n0 + g * 16 + ml) * DM;

  for (int kk = 0; kk < DM; kk += 32) {
    bf16x16 af = load_a_frag(arow + kk, half);
#pragma unroll
    for (int g = 0; g < 4; ++g)
      acc[g] = WMMA_BF16(af, load_b_frag(bcol[g] + kk, half), acc[g]);
  }
#pragma unroll
  for (int g = 0; g < 4; ++g)
#pragma unroll
    for (int i = 0; i < 8; ++i) {
      int m = m0 + i + half * 8;
      int n = n0 + g * 16 + ml;
      int b = m >> 11, s = m & (SEQ - 1);
      int h = n >> 6,  d = n & (HD - 1);
      size_t idx = (mode == 2)
          ? ((size_t)(b * NH + h) * HD + d) * SEQ + s
          : ((size_t)(b * NH + h) * SEQ + s) * HD + d;
      dst[idx] = (__bf16)acc[g][i];
    }
}

// ---------------- flash attention: one wave per (b,h,16-row q-tile) ----------------
__global__ __launch_bounds__(32) void attn_kernel(
    const __bf16* __restrict__ Q, const __bf16* __restrict__ K,
    const __bf16* __restrict__ Vt, const float* __restrict__ lut,
    __bf16* __restrict__ ctx) {
  const int lane = threadIdx.x, half = lane >> 4, ml = lane & 15;
  const int q0 = blockIdx.x * 16, h = blockIdx.y, b = blockIdx.z;
  const size_t bh = (size_t)b * NH + h;

  __shared__ __bf16 sP[16 * 32];   // P tile staging: D-layout -> A-layout transpose

  const __bf16* qrow = Q + (bh * SEQ + q0 + ml) * HD;
  bf16x16 aq0 = load_a_frag(qrow, half);        // d = 0..31
  bf16x16 aq1 = load_a_frag(qrow + 32, half);   // d = 32..63

  f32x8 acc[4] = {};           // context accumulator, 16 x 64, D-layout
  float rmax[8], rsum[8];      // per-row stats (rows half*8+i), uniform across 16-lane half
#pragma unroll
  for (int i = 0; i < 8; ++i) { rmax[i] = -1e30f; rsum[i] = 0.f; }

  const float* lrow = lut + (size_t)h * 4096 + 2047 - (q0 + half * 8);

  for (int kc = 0; kc < SEQ; kc += 32) {
    // S[16x32] = Q[16x64] @ K_chunk^T[64x32]
    f32x8 s0 = {}, s1 = {};
    const __bf16* kc0 = K + (bh * SEQ + kc + ml) * HD;        // cols kc..kc+15
    const __bf16* kc1 = K + (bh * SEQ + kc + 16 + ml) * HD;   // cols kc+16..kc+31
    s0 = WMMA_BF16(aq0, load_b_frag(kc0,      half), s0);
    s0 = WMMA_BF16(aq1, load_b_frag(kc0 + 32, half), s0);
    s1 = WMMA_BF16(aq0, load_b_frag(kc1,      half), s1);
    s1 = WMMA_BF16(aq1, load_b_frag(kc1 + 32, half), s1);

    // + position bias, online softmax update, stage P to LDS as bf16
#pragma unroll
    for (int i = 0; i < 8; ++i) {
      s0[i] += lrow[kc + ml - i];
      s1[i] += lrow[kc + 16 + ml - i];
      float cm = fmaxf(s0[i], s1[i]);
#pragma unroll
      for (int off = 8; off >= 1; off >>= 1) cm = fmaxf(cm, __shfl_xor(cm, off, 32));
      float nm = fmaxf(rmax[i], cm);
      float sc = __expf(rmax[i] - nm);
      rmax[i] = nm;
      float e0 = __expf(s0[i] - nm);
      float e1 = __expf(s1[i] - nm);
      float rs = e0 + e1;
#pragma unroll
      for (int off = 8; off >= 1; off >>= 1) rs += __shfl_xor(rs, off, 32);
      rsum[i] = rsum[i] * sc + rs;
      acc[0][i] *= sc; acc[1][i] *= sc; acc[2][i] *= sc; acc[3][i] *= sc;
      sP[(i + half * 8) * 32 + ml]      = (__bf16)e0;
      sP[(i + half * 8) * 32 + 16 + ml] = (__bf16)e1;
    }
    asm volatile("s_wait_dscnt 0x0" ::: "memory");   // single wave: DS in-order, fence only

    // re-load P in A-layout
    bf16x16 ap;
    {
      const __bf16* pr = &sP[ml * 32 + half * 8];
      bf16x8 lo = *(const bf16x8*)(pr);
      bf16x8 hi = *(const bf16x8*)(pr + 16);
#pragma unroll
      for (int j = 0; j < 8; ++j) { ap[j] = lo[j]; ap[j + 8] = hi[j]; }
    }
    // context[16x64] += P[16x32] @ V_chunk[32x64]  (Vt gives contiguous B runs)
#pragma unroll
    for (int g = 0; g < 4; ++g) {
      const __bf16* vrow = Vt + (bh * HD + g * 16 + ml) * SEQ + kc;
      acc[g] = WMMA_BF16(ap, load_b_frag(vrow, half), acc[g]);
    }
  }

  // normalize and emit ctx as bf16 [b*2048+s][h*64+d] (A operand of output GEMM)
#pragma unroll
  for (int g = 0; g < 4; ++g)
#pragma unroll
    for (int i = 0; i < 8; ++i) {
      int q = q0 + i + half * 8;
      float v = acc[g][i] / rsum[i];
      ctx[((size_t)(b * SEQ + q)) * DM + h * HD + g * 16 + ml] = (__bf16)v;
    }
}

// ---------------- output projection: ctx[4096x1024] @ woT -> out (f32) ----------------
__global__ __launch_bounds__(32) void gemm_out_kernel(
    const __bf16* __restrict__ A, const __bf16* __restrict__ BT,
    float* __restrict__ dst) {
  const int lane = threadIdx.x, half = lane >> 4, ml = lane & 15;
  const int m0 = blockIdx.x * 16, n0 = blockIdx.y * 64;

  f32x8 acc[4] = {};
  const __bf16* arow = A + (size_t)(m0 + ml) * DM;
  const __bf16* bcol[4];
#pragma unroll
  for (int g = 0; g < 4; ++g) bcol[g] = BT + (size_t)(n0 + g * 16 + ml) * DM;

  for (int kk = 0; kk < DM; kk += 32) {
    bf16x16 af = load_a_frag(arow + kk, half);
#pragma unroll
    for (int g = 0; g < 4; ++g)
      acc[g] = WMMA_BF16(af, load_b_frag(bcol[g] + kk, half), acc[g]);
  }
#pragma unroll
  for (int g = 0; g < 4; ++g)
#pragma unroll
    for (int i = 0; i < 8; ++i) {
      int m = m0 + i + half * 8;
      int n = n0 + g * 16 + ml;
      dst[(size_t)m * DM + n] = acc[g][i];
    }
}

// ---------------- workspace layout (bytes) ----------------
static constexpr size_t SZ_XB  = (size_t)NB * SEQ * DM * 2;      // 8 MB
static constexpr size_t SZ_W   = (size_t)DM * DM * 2;            // 2 MB each
static constexpr size_t SZ_QKV = (size_t)NB * NH * SEQ * HD * 2; // 8 MB each
static constexpr size_t OFF_XB  = 0;
static constexpr size_t OFF_WQT = OFF_XB  + SZ_XB;
static constexpr size_t OFF_WKT = OFF_WQT + SZ_W;
static constexpr size_t OFF_WVT = OFF_WKT + SZ_W;
static constexpr size_t OFF_WOT = OFF_WVT + SZ_W;
static constexpr size_t OFF_Q   = OFF_WOT + SZ_W;
static constexpr size_t OFF_K   = OFF_Q   + SZ_QKV;
static constexpr size_t OFF_VT  = OFF_K   + SZ_QKV;
static constexpr size_t OFF_CTX = OFF_VT  + SZ_QKV;
static constexpr size_t OFF_LUT = OFF_CTX + SZ_QKV;

extern "C" void kernel_launch(void* const* d_in, const int* in_sizes, int n_in,
                              void* d_out, int out_size, void* d_ws, size_t ws_size,
                              hipStream_t stream) {
  (void)in_sizes; (void)n_in; (void)out_size; (void)ws_size;
  const float* x   = (const float*)d_in[0];
  const float* wq  = (const float*)d_in[1];
  const float* wk  = (const float*)d_in[2];
  const float* wv  = (const float*)d_in[3];
  const float* wo  = (const float*)d_in[4];
  const float* rel = (const float*)d_in[5];

  float* out      = (float*)d_out;
  float* bias_out = out + (size_t)NB * SEQ * DM;

  char* ws = (char*)d_ws;
  __bf16* xb  = (__bf16*)(ws + OFF_XB);
  __bf16* wqT = (__bf16*)(ws + OFF_WQT);
  __bf16* wkT = (__bf16*)(ws + OFF_WKT);
  __bf16* wvT = (__bf16*)(ws + OFF_WVT);
  __bf16* woT = (__bf16*)(ws + OFF_WOT);
  __bf16* Qb  = (__bf16*)(ws + OFF_Q);
  __bf16* Kb  = (__bf16*)(ws + OFF_K);
  __bf16* Vt  = (__bf16*)(ws + OFF_VT);
  __bf16* ctx = (__bf16*)(ws + OFF_CTX);
  float*  lut = (float*)(ws + OFF_LUT);

  // 1) bias LUT + bias output (pure streaming store, LUT lives in L2)
  build_lut_kernel<<<(NH * 4096 + 255) / 256, 256, 0, stream>>>(rel, lut);
  write_bias_kernel<<<(NH * SEQ * SEQ / 4) / 256, 256, 0, stream>>>(lut, bias_out);

  // 2) weight transposes + x down-convert
  transpose_w_kernel<<<(DM * DM) / 256, 256, 0, stream>>>(wq, wqT);
  transpose_w_kernel<<<(DM * DM) / 256, 256, 0, stream>>>(wk, wkT);
  transpose_w_kernel<<<(DM * DM) / 256, 256, 0, stream>>>(wv, wvT);
  transpose_w_kernel<<<(DM * DM) / 256, 256, 0, stream>>>(wo, woT);
  cvt_x_kernel<<<(NB * SEQ * DM / 4 + 255) / 256, 256, 0, stream>>>(
      x, xb, NB * SEQ * DM / 4);

  // 3) Q/K/V projections (bf16 WMMA)
  dim3 gg(NB * SEQ / 16, DM / 64);
  gemm_qkv_kernel<<<gg, 32, 0, stream>>>(xb, wqT, Qb, 0);
  gemm_qkv_kernel<<<gg, 32, 0, stream>>>(xb, wkT, Kb, 1);
  gemm_qkv_kernel<<<gg, 32, 0, stream>>>(xb, wvT, Vt, 2);

  // 4) flash attention with fused position bias
  attn_kernel<<<dim3(SEQ / 16, NH, NB), 32, 0, stream>>>(Qb, Kb, Vt, lut, ctx);

  // 5) output projection (f32 store)
  gemm_out_kernel<<<gg, 32, 0, stream>>>(ctx, woT, out);
}
